// VectorQuantizer_61658550502008
// MI455X (gfx1250) — matrix operand order repaired and checked by reading, not compiled, not run on hardware
//
#include <hip/hip_runtime.h>
#include <math.h>

// ---- problem constants (from reference) ----
#define EMB_DIM   64
#define NUM_CODES 1024
#define M_TILES   2                      // 16x16 M-tiles per block
#define ROWS_PER_BLOCK (M_TILES * 16)    // 32 input rows per block
#define THREADS   256                    // 8 waves (wave32)
#define N_WAVES   8
#define COL_TILES_PER_WAVE 8             // 64 col tiles / 8 waves

typedef __attribute__((ext_vector_type(16))) _Float16 v16h;
typedef __attribute__((ext_vector_type(8)))  _Float16 v8h;
typedef __attribute__((ext_vector_type(4)))  _Float16 v4h;
typedef __attribute__((ext_vector_type(8)))  float    v8f;

static __device__ __forceinline__ v16h combine8(v8h lo, v8h hi) {
    v16h r;
#pragma unroll
    for (int i = 0; i < 8; ++i) { r[i] = lo[i]; r[i + 8] = hi[i]; }
    return r;
}

static __device__ __forceinline__ unsigned int umax2(unsigned int a, unsigned int b) {
    return a > b ? a : b;
}

// ---------------- kernel 0: zero loss scalar + histogram ----------------
__global__ void vq_init_kernel(float* __restrict__ out, unsigned int* __restrict__ counts) {
    int t = blockIdx.x * blockDim.x + threadIdx.x;
    if (t < NUM_CODES) counts[t] = 0u;
    if (t == 0) out[0] = 0.0f;
}

// ------------- kernel 1: normalize codebook, make f16 copy --------------
__global__ __launch_bounds__(THREADS)
void vq_prep_weights(const float* __restrict__ w,
                     float* __restrict__ wf32,
                     _Float16* __restrict__ wf16,
                     float* __restrict__ wnorm2) {
    const int wave = threadIdx.x >> 5;
    const int lane = threadIdx.x & 31;
    const int k = blockIdx.x * N_WAVES + wave;          // code row

    const float2 v = *(const float2*)(w + (size_t)k * EMB_DIM + lane * 2);
    float ss = v.x * v.x + v.y * v.y;
#pragma unroll
    for (int m = 1; m < 32; m <<= 1) ss += __shfl_xor(ss, m, 32);

    const float scale = 1.0f / fmaxf(sqrtf(ss), 1e-12f);
    const float a = v.x * scale, b = v.y * scale;

    *(float2*)(wf32 + (size_t)k * EMB_DIM + lane * 2) = make_float2(a, b);
    _Float16* o16 = wf16 + (size_t)k * EMB_DIM + lane * 2;
    o16[0] = (_Float16)a; o16[1] = (_Float16)b;
    if (lane == 0) wnorm2[k] = ss * scale * scale;
}

// --------------------------- kernel 2: main -----------------------------
// One block = 32 input rows (2 WMMA M-tiles). 8 waves split 64 column tiles;
// each B fragment pair feeds 4 WMMAs (2 M-tiles x K=64).
// Bias (2 - 0.5||w||^2) is preloaded into the WMMA C operand, so the
// accumulator exits the WMMA already positive and the argmax is a pure
// 32-bit key max:  key = (bits(acc) & ~1023) | (1023 - code).
__global__ __launch_bounds__(THREADS)
void vq_main(const float* __restrict__ in,
             const float* __restrict__ wf32,
             const _Float16* __restrict__ wf16,
             const float* __restrict__ wnorm2,
             unsigned int* __restrict__ counts,
             float* __restrict__ out,
             int quantOff, int idxOff, float lossScale) {
    __shared__ float    xf32s[ROWS_PER_BLOCK * EMB_DIM];                  // 8 KB
    __shared__ _Float16 xf16s[ROWS_PER_BLOCK * EMB_DIM];                  // 4 KB
    __shared__ unsigned int bestLds[N_WAVES * ROWS_PER_BLOCK * 16];       // 16 KB
    __shared__ int      rowIdxS[ROWS_PER_BLOCK];
    __shared__ float    lossPart[N_WAVES];

    const int tid  = threadIdx.x;
    const int wave = tid >> 5;
    const int lane = tid & 31;
    const int nIdx = lane & 15;     // column-within-tile
    const int hi   = lane >> 4;     // 0: lanes 0-15, 1: lanes 16-31
    const int gRowBase = blockIdx.x * ROWS_PER_BLOCK;

    // ---- phase 1: L2-normalize 32 rows into LDS (f32 + f16) ----
    {
        const int row = tid >> 3;          // 0..31
        const int j   = tid & 7;           // 8 threads/row, 8 elems each
        const float* src = in + (size_t)(gRowBase + row) * EMB_DIM + j * 8;
        const float4 x0 = *(const float4*)src;
        const float4 x1 = *(const float4*)(src + 4);
        float ss = x0.x * x0.x + x0.y * x0.y + x0.z * x0.z + x0.w * x0.w
                 + x1.x * x1.x + x1.y * x1.y + x1.z * x1.z + x1.w * x1.w;
#pragma unroll
        for (int m = 1; m < 8; m <<= 1) ss += __shfl_xor(ss, m, 8);  // row = one 8-lane group
        const float scale = 1.0f / fmaxf(sqrtf(ss), 1e-12f);
        const float4 n0 = make_float4(x0.x * scale, x0.y * scale, x0.z * scale, x0.w * scale);
        const float4 n1 = make_float4(x1.x * scale, x1.y * scale, x1.z * scale, x1.w * scale);
        *(float4*)&xf32s[row * EMB_DIM + j * 8]     = n0;
        *(float4*)&xf32s[row * EMB_DIM + j * 8 + 4] = n1;
        v4h h0; h0[0] = (_Float16)n0.x; h0[1] = (_Float16)n0.y;
                h0[2] = (_Float16)n0.z; h0[3] = (_Float16)n0.w;
        v4h h1; h1[0] = (_Float16)n1.x; h1[1] = (_Float16)n1.y;
                h1[2] = (_Float16)n1.z; h1[3] = (_Float16)n1.w;
        *(v4h*)&xf16s[row * EMB_DIM + j * 8]     = h0;
        *(v4h*)&xf16s[row * EMB_DIM + j * 8 + 4] = h1;
    }
    __syncthreads();

    // ---- phase 2: WMMA similarity + per-lane keyed argmax ----
    // A fragment (16x32 f16), contiguous layout: lane<16 -> K {0..15}; lane>=16 -> K {16..31}
    v16h a[M_TILES][2];
#pragma unroll
    for (int mt = 0; mt < M_TILES; ++mt) {
#pragma unroll
        for (int kk = 0; kk < 2; ++kk) {
            const int base = (mt * 16 + nIdx) * EMB_DIM + kk * 32 + hi * 16;
            a[mt][kk] = combine8(*(const v8h*)&xf16s[base], *(const v8h*)&xf16s[base + 8]);
        }
    }

    unsigned int bestK[M_TILES][8];
#pragma unroll
    for (int mt = 0; mt < M_TILES; ++mt)
#pragma unroll
        for (int r = 0; r < 8; ++r) bestK[mt][r] = 0u;

    // single base pointer; all tile addressing becomes compile-time IOFFSETs
    const int code0 = wave * COL_TILES_PER_WAVE * 16 + nIdx;   // this lane's first code
    const _Float16* bbase  = wf16 + (size_t)code0 * EMB_DIM + hi * 16;
    const float*    wnbase = wnorm2 + code0;
    const unsigned int invBase = (unsigned int)(NUM_CODES - 1 - code0);

#pragma unroll
    for (int t = 0; t < COL_TILES_PER_WAVE; ++t) {
        const int toff = t * 16 * EMB_DIM;                     // +t*2048 bytes (imm offset)
        const v16h b0 = combine8(*(const v8h*)(bbase + toff),
                                 *(const v8h*)(bbase + toff + 8));
        const v16h b1 = combine8(*(const v8h*)(bbase + toff + 32),
                                 *(const v8h*)(bbase + toff + 40));
        const float bias = fmaf(-0.5f, wnbase[t * 16], 2.0f);  // keeps score' > 0
        const unsigned int invCode = invBase - (unsigned int)(t * 16);
        v8f cinit;
#pragma unroll
        for (int r = 0; r < 8; ++r) cinit[r] = bias;           // bias folded into WMMA C
#pragma unroll
        for (int mt = 0; mt < M_TILES; ++mt) {
            v8f acc = __builtin_amdgcn_wmma_f32_16x16x32_f16(false, a[mt][0], false, b0,
                                                             (short)0, cinit, false, false);
            acc = __builtin_amdgcn_wmma_f32_16x16x32_f16(false, a[mt][1], false, b1,
                                                         (short)0, acc, false, false);
#pragma unroll
            for (int r = 0; r < 8; ++r) {                      // row = mt*16 + hi*8 + r
                const unsigned int key =
                    (__float_as_uint(acc[r]) & 0xFFFFFC00u) | invCode;
                bestK[mt][r] = umax2(bestK[mt][r], key);
            }
        }
    }

    // dump per-lane winners to LDS (32-bit keys; no shuffles, no branches)
#pragma unroll
    for (int mt = 0; mt < M_TILES; ++mt) {
#pragma unroll
        for (int r = 0; r < 8; ++r) {
            const int row = mt * 16 + hi * 8 + r;
            bestLds[((wave * ROWS_PER_BLOCK + row) << 4) + nIdx] = bestK[mt][r];
        }
    }
    __syncthreads();

    // ---- phase 3: reduce 8 waves x 16 lanes per row; emit index + histogram ----
    {
        const int row = tid >> 3;          // 0..31
        const int j   = tid & 7;           // 8 threads per row
        const unsigned int* p = &bestLds[row << 4];
        unsigned int k = 0u;
#pragma unroll
        for (int w2 = 0; w2 < N_WAVES; ++w2) {
            const int o = w2 * (ROWS_PER_BLOCK * 16);          // constant DS offsets
            k = umax2(k, umax2(p[o + j], p[o + j + 8]));
        }
#pragma unroll
        for (int msk = 1; msk < 8; msk <<= 1) {
            k = umax2(k, (unsigned int)__shfl_xor((int)k, msk, 8));
        }
        if (j == 0) {
            const int code = (NUM_CODES - 1) - (int)(k & (NUM_CODES - 1));
            rowIdxS[row] = code;
            atomicAdd(&counts[code], 1u);
            out[idxOff + gRowBase + row] = (float)code;   // indices as f32 stream
        }
    }
    __syncthreads();

    // ---- phase 4: quantized output (== STE forward) + loss partial ----
    {
        const int row  = tid >> 3;
        const int j    = tid & 7;
        const int code = rowIdxS[row];
        const float* qp = wf32 + (size_t)code * EMB_DIM + j * 8;
        const float4 q0 = *(const float4*)qp;
        const float4 q1 = *(const float4*)(qp + 4);
        const float4 x0 = *(const float4*)&xf32s[row * EMB_DIM + j * 8];
        const float4 x1 = *(const float4*)&xf32s[row * EMB_DIM + j * 8 + 4];
        float* o = out + quantOff + (size_t)(gRowBase + row) * EMB_DIM + j * 8;
        o[0] = q0.x; o[1] = q0.y; o[2] = q0.z; o[3] = q0.w;   // out+1 base => scalar stores
        o[4] = q1.x; o[5] = q1.y; o[6] = q1.z; o[7] = q1.w;
        float part = (q0.x - x0.x) * (q0.x - x0.x) + (q0.y - x0.y) * (q0.y - x0.y)
                   + (q0.z - x0.z) * (q0.z - x0.z) + (q0.w - x0.w) * (q0.w - x0.w)
                   + (q1.x - x1.x) * (q1.x - x1.x) + (q1.y - x1.y) * (q1.y - x1.y)
                   + (q1.z - x1.z) * (q1.z - x1.z) + (q1.w - x1.w) * (q1.w - x1.w);
#pragma unroll
        for (int msk = 1; msk < 32; msk <<= 1) part += __shfl_xor(part, msk, 32);
        if (lane == 0) lossPart[wave] = part;
    }
    __syncthreads();
    if (tid == 0) {
        float s = 0.0f;
#pragma unroll
        for (int w2 = 0; w2 < N_WAVES; ++w2) s += lossPart[w2];
        atomicAdd(out, s * lossScale);   // lossScale = 1.25 / numel
    }
}

// ----------------------- kernel 3: perplexity ---------------------------
__global__ __launch_bounds__(1024)
void vq_perplexity(const unsigned int* __restrict__ counts,
                   float* __restrict__ perpOut, float invN) {
    __shared__ float red[32];
    const int t = threadIdx.x;
    const float p = (float)counts[t] * invN;
    float term = p * logf(p + 1e-10f);
#pragma unroll
    for (int msk = 1; msk < 32; msk <<= 1) term += __shfl_xor(term, msk, 32);
    if ((t & 31) == 0) red[t >> 5] = term;
    __syncthreads();
    if (t < 32) {
        float v = red[t];
#pragma unroll
        for (int msk = 1; msk < 32; msk <<= 1) v += __shfl_xor(v, msk, 32);
        if (t == 0) *perpOut = expf(-v);
    }
}

// ------------------------------ launcher --------------------------------
extern "C" void kernel_launch(void* const* d_in, const int* in_sizes, int n_in,
                              void* d_out, int out_size, void* d_ws, size_t ws_size,
                              hipStream_t stream) {
    const float* in     = (const float*)d_in[0];   // [16,4096,64] f32
    const float* weight = (const float*)d_in[1];   // [1024,64] f32
    float* out = (float*)d_out;

    char* ws = (char*)d_ws;
    float*        wf32   = (float*)ws;                                   // 256 KB
    _Float16*     wf16   = (_Float16*)(ws + 262144);                     // 128 KB
    float*        wnorm2 = (float*)(ws + 262144 + 131072);               //   4 KB
    unsigned int* counts = (unsigned int*)(ws + 262144 + 131072 + 4096); //   4 KB

    const int nElem = in_sizes[0];            // 4194304
    const int nVec  = nElem / EMB_DIM;        // 65536
    const int quantOff = 1;
    const int perpOff  = 1 + nElem;
    const int idxOff   = perpOff + 1;
    const float lossScale = 1.25f / (float)nElem;

    vq_init_kernel<<<4, 256, 0, stream>>>(out, counts);
    vq_prep_weights<<<NUM_CODES / N_WAVES, THREADS, 0, stream>>>(weight, wf32, wf16, wnorm2);
    vq_main<<<nVec / ROWS_PER_BLOCK, THREADS, 0, stream>>>(
        in, wf32, wf16, wnorm2, counts, out, quantOff, idxOff, lossScale);
    vq_perplexity<<<1, 1024, 0, stream>>>(counts, out + perpOff, 1.0f / (float)nVec);
}
